// FieldEvolution_63651415327278
// MI455X (gfx1250) — matrix operand order repaired and checked by reading, not compiled, not run on hardware
//
#include <hip/hip_runtime.h>

// ---------------- problem constants (from setup_inputs) ----------------
constexpr int kB = 4;        // batch
constexpr int kS = 4096;     // seq
constexpr int kD = 256;      // dim
constexpr int kBM = 128;     // i-rows per workgroup
constexpr int kBN = 32;      // j-cols per block
constexpr int kNW = 8;       // waves per workgroup (256 threads, wave32)
constexpr int kFJ = 132;     // LDS fj row stride in uints (264 bf16, padded)
constexpr int kFT = 20;      // LDS fjT col stride in uints (16 used, padded, 16B-aligned)
constexpr int kSS = 20;      // LDS S row stride in uints (16 used, padded, 16B-aligned)
constexpr float kDT = 0.01f;
constexpr int kSTEPS = 10;

typedef __attribute__((ext_vector_type(16))) __bf16 v16bf;
typedef __attribute__((ext_vector_type(8)))  float  v8f;

union FragU {
    unsigned int u[8];
    uint4 q[2];
    v16bf bf;
};

__device__ __forceinline__ unsigned short f2bf(float f) {
    union { float f; unsigned int u; } c; c.f = f;
    unsigned int u = c.u + 0x7FFFu + ((c.u >> 16) & 1u);  // RNE
    return (unsigned short)(u >> 16);
}

// ---------------- prep: fp32 field -> bf16 copy + per-row sum of squares ----
__global__ void fe_prep(const float* __restrict__ f,
                        unsigned short* __restrict__ fb,
                        float* __restrict__ sq) {
    const int tid = threadIdx.x;
    const int lane = tid & 31;
    const int row = blockIdx.x * 8 + (tid >> 5);
    const float* src = f + (size_t)row * kD;
    float vals[8];
    float s = 0.f;
#pragma unroll
    for (int t = 0; t < 8; ++t) {
        float v = src[lane + 32 * t];
        vals[t] = v;
        s += v * v;
    }
#pragma unroll
    for (int w = 16; w >= 1; w >>= 1) s += __shfl_xor(s, w, 32);
    unsigned short* dst = fb + (size_t)row * kD;
#pragma unroll
    for (int t = 0; t < 8; ++t) dst[lane + 32 * t] = f2bf(vals[t]);
    if (lane == 0) sq[row] = s;
}

// ---------------- fused force + leapfrog update (+optional normalize) ------
// out = base + scale * (G(op) @ op);  G = -g/sqrt(sqi+sqj-2*op.op^T+1e-8) * tril
__global__ __launch_bounds__(256) void fe_force(
    const float* __restrict__ base,          // fp32 field for the update term
    const unsigned short* __restrict__ fbf,  // bf16 operand field
    const float* __restrict__ sqg,           // per-row sum of squares of operand
    const float* __restrict__ gptr,
    float scale, int do_norm,
    float* __restrict__ outf,
    unsigned short* __restrict__ outbf,
    float* __restrict__ outsq) {
    // double-buffered staging tiles (one barrier per j-block)
    __shared__ unsigned int lds_fj[2][kBN * kFJ];    // row-major bf16 tile
    __shared__ unsigned int lds_fjT[2][kD * kFT];    // pair-packed "transpose" (k, k+16)
    __shared__ float lds_sqj[2][kBN];
    __shared__ float lds_sqi[kBM];
    __shared__ unsigned int lds_S[kNW * 16 * kSS];   // per-wave S tiles (pair-packed)

    const int tid = threadIdx.x;
    const int lane = tid & 31;
    const int wave = tid >> 5;
    const int n16 = lane & 15;
    const int kh = lane >> 4;
    const int batch = blockIdx.y;
    const int ib = blockIdx.x * kBM;
    const float g = gptr[0];
    const size_t fb0 = (size_t)batch * kS * kD;

    if (tid < kBM) lds_sqi[tid] = sqg[batch * kS + ib + tid];

    // A fragments: this wave's 16 rows of operand field, all 8 K-steps, in regs.
    const int irow = ib + wave * 16 + n16;
    FragU afrag[8];
    {
        const unsigned int* ar = (const unsigned int*)(fbf + fb0 + (size_t)irow * kD);
#pragma unroll
        for (int t = 0; t < 8; ++t) {
            afrag[t].q[0] = *(const uint4*)(ar + 16 * t + 4 * kh);
            afrag[t].q[1] = *(const uint4*)(ar + 16 * t + 8 + 4 * kh);
        }
    }

    const v8f vzero = {0.f, 0.f, 0.f, 0.f, 0.f, 0.f, 0.f, 0.f};
    v8f facc[16];
#pragma unroll
    for (int i = 0; i < 16; ++i) facc[i] = vzero;

    const int iwave_max = ib + wave * 16 + 15;
    const int njb = (ib + kBM) / kBN;
    unsigned int* myS = lds_S + wave * 16 * kSS;

    // ---- staging helpers: global -> regs -> (after compute) LDS ----
    const int r   = tid >> 3;          // row-major: row
    const int c0r = (tid & 7) * 16;    //            uint col start
    const int p   = tid & 15;          // transpose: pair id (distinct banks/half-wave)
    const int c0t = (tid >> 4) * 16;   //            column chunk
    uint4 pr[4];                        // staged row-major piece
    union PK { uint4 q[2]; unsigned int u[8]; } pa, pb;  // staged transpose rows
    float psq = 0.f;

    auto load_next = [&](int jb2) {
        const uint4* src4 =
            (const uint4*)((const unsigned int*)(fbf + fb0 + (size_t)(jb2 + r) * kD) + c0r);
#pragma unroll
        for (int c = 0; c < 4; ++c) pr[c] = src4[c];
        const unsigned int* rowA =
            (const unsigned int*)(fbf + fb0 + (size_t)(jb2 + p) * kD) + (c0t >> 1);
        const unsigned int* rowB =
            (const unsigned int*)(fbf + fb0 + (size_t)(jb2 + p + 16) * kD) + (c0t >> 1);
        pa.q[0] = *(const uint4*)(rowA);  pa.q[1] = *(const uint4*)(rowA + 4);
        pb.q[0] = *(const uint4*)(rowB);  pb.q[1] = *(const uint4*)(rowB + 4);
        if (tid < kBN) psq = sqg[batch * kS + jb2 + tid];
    };
    auto store_next = [&](int buf) {
        uint4* dst4 = (uint4*)(lds_fj[buf] + r * kFJ + c0r);
#pragma unroll
        for (int c = 0; c < 4; ++c) dst4[c] = pr[c];
#pragma unroll
        for (int c = 0; c < 16; ++c) {
            unsigned int a = pa.u[c >> 1], b = pb.u[c >> 1];
            // pack bf16(rowA@col) | bf16(rowB@col) << 16 via v_perm
            unsigned int val = (c & 1) ? __builtin_amdgcn_perm(b, a, 0x07060302u)
                                       : __builtin_amdgcn_perm(b, a, 0x05040100u);
            lds_fjT[buf][(c0t + c) * kFT + p] = val;
        }
        if (tid < kBN) lds_sqj[buf][tid] = psq;
    };

    // prologue: stage j-block 0
    load_next(0);
    store_next(0);
    __syncthreads();

    for (int jblk = 0; jblk < njb; ++jblk) {
        const int jb = jblk * kBN;
        const int cur = jblk & 1;
        const bool have_next = (jblk + 1 < njb);
        if (have_next) load_next(jb + kBN);   // vmem in flight during compute

        if (jb <= iwave_max) {
            const unsigned int* fjc = lds_fj[cur];
            const unsigned int* fjt = lds_fjT[cur];

            // ---- matmul 1: dot = field_i . field_j^T (K over d), pipelined ----
            v8f acc[2];
            acc[0] = vzero; acc[1] = vzero;
            FragU bb[2];
            auto loadB1 = [&](int idx, FragU& dst) {
                const int t = idx >> 1, nt = idx & 1;
                const unsigned int* br = fjc + (nt * 16 + n16) * kFJ + 16 * t + 4 * kh;
                dst.q[0] = *(const uint4*)(br);
                dst.q[1] = *(const uint4*)(br + 8);
            };
            loadB1(0, bb[0]);
#pragma unroll
            for (int idx = 0; idx < 16; ++idx) {
                if (idx < 15) loadB1(idx + 1, bb[(idx + 1) & 1]);
                const int t = idx >> 1, nt = idx & 1;
                acc[nt] = __builtin_amdgcn_wmma_f32_16x16x32_bf16(
                    false, afrag[t].bf, false, bb[idx & 1].bf, (short)0, acc[nt], false, false);
            }

            // ---- elementwise: S = tril * (-g * rsqrt(d2)) -> packed bf16 pairs ----
            // C layout: lane = N, row m = v + 8*kh; pack (nt=0, nt=1) into one uint
#pragma unroll
            for (int v = 0; v < 8; ++v) {
                int m = v + 8 * kh;
                int i = ib + wave * 16 + m;
                float sqi = lds_sqi[wave * 16 + m];
                int j0 = jb + n16;
                int j1 = jb + 16 + n16;
                float d20 = sqi + lds_sqj[cur][n16]      - 2.0f * acc[0][v] + 1e-8f;
                float d21 = sqi + lds_sqj[cur][16 + n16] - 2.0f * acc[1][v] + 1e-8f;
                float v0 = (j0 <= i) ? (-g * rsqrtf(d20)) : 0.0f;
                float v1 = (j1 <= i) ? (-g * rsqrtf(d21)) : 0.0f;
                myS[m * kSS + n16] = (unsigned int)f2bf(v0) | ((unsigned int)f2bf(v1) << 16);
            }

            // ---- re-read S in A layout (K permuted as pairs (n, n+16)) ----
            FragU a2;
            a2.q[0] = *(const uint4*)(myS + n16 * kSS + 8 * kh);
            a2.q[1] = *(const uint4*)(myS + n16 * kSS + 4 + 8 * kh);

            // ---- matmul 2: force += S . field_j (same K permutation), pipelined ----
            FragU b2[2];
            auto loadB2 = [&](int nt2, FragU& dst) {
                const unsigned int* bt = fjt + (nt2 * 16 + n16) * kFT + 8 * kh;
                dst.q[0] = *(const uint4*)(bt);
                dst.q[1] = *(const uint4*)(bt + 4);
            };
            loadB2(0, b2[0]);
#pragma unroll
            for (int nt2 = 0; nt2 < 16; ++nt2) {
                if (nt2 < 15) loadB2(nt2 + 1, b2[(nt2 + 1) & 1]);
                facc[nt2] = __builtin_amdgcn_wmma_f32_16x16x32_bf16(
                    false, a2.bf, false, b2[nt2 & 1].bf, (short)0, facc[nt2], false, false);
            }
        }

        if (have_next) store_next(cur ^ 1);   // regs -> other LDS buffer
        __syncthreads();
    }

    // ---- epilogue: out = base + scale*force; row norm; emit f32/bf16/sq ----
    {
        const float* bp = base + fb0;
        float rowsum[8];
#pragma unroll
        for (int v = 0; v < 8; ++v) rowsum[v] = 0.f;
#pragma unroll
        for (int nt2 = 0; nt2 < 16; ++nt2)
#pragma unroll
            for (int v = 0; v < 8; ++v) {
                int m = v + 8 * kh;
                int row = ib + wave * 16 + m;
                int col = nt2 * 16 + n16;
                float o = bp[(size_t)row * kD + col] + scale * facc[nt2][v];
                facc[nt2][v] = o;
                rowsum[v] += o * o;
            }
#pragma unroll
        for (int v = 0; v < 8; ++v)
#pragma unroll
            for (int w = 8; w >= 1; w >>= 1) rowsum[v] += __shfl_xor(rowsum[v], w, 32);
#pragma unroll
        for (int v = 0; v < 8; ++v) {
            float s = rowsum[v];
            float inv = 1.0f;
            if (do_norm) { inv = 1.0f / (sqrtf(s) + 1e-8f); s = s * inv * inv; }
            int m = v + 8 * kh;
            int row = ib + wave * 16 + m;
            if (n16 == 0) outsq[batch * kS + row] = s;
#pragma unroll
            for (int nt2 = 0; nt2 < 16; ++nt2) {
                int col = nt2 * 16 + n16;
                float o = facc[nt2][v] * inv;
                outf[fb0 + (size_t)row * kD + col] = o;
                outbf[fb0 + (size_t)row * kD + col] = f2bf(o);
            }
        }
    }
}

extern "C" void kernel_launch(void* const* d_in, const int* in_sizes, int n_in,
                              void* d_out, int out_size, void* d_ws, size_t ws_size,
                              hipStream_t stream) {
    (void)in_sizes; (void)n_in; (void)out_size; (void)ws_size;
    const float* field_in = (const float*)d_in[0];
    const float* gptr = (const float*)d_in[1];

    const size_t N = (size_t)kB * kS * kD;
    char* ws = (char*)d_ws;
    float* bufA = (float*)ws;                  ws += N * sizeof(float);
    float* bufH = (float*)ws;                  ws += N * sizeof(float);
    unsigned short* bfA = (unsigned short*)ws; ws += N * sizeof(unsigned short);
    unsigned short* bfH = (unsigned short*)ws; ws += N * sizeof(unsigned short);
    float* sqA = (float*)ws;                   ws += (size_t)kB * kS * sizeof(float);
    float* sqH = (float*)ws;

    dim3 fgrid(kS / kBM, kB);

    fe_prep<<<(kB * kS) / 8, 256, 0, stream>>>(field_in, bfA, sqA);

    const float* cur = field_in;
    for (int step = 0; step < kSTEPS; ++step) {
        // half step: field_half = field + 0.5*dt*force(field)
        fe_force<<<fgrid, 256, 0, stream>>>(cur, bfA, sqA, gptr,
                                            0.5f * kDT, 0, bufH, bfH, sqH);
        // full step: field = normalize(field + dt*force(field_half))
        float* outfield = (step == kSTEPS - 1) ? (float*)d_out : bufA;
        fe_force<<<fgrid, 256, 0, stream>>>(cur, bfH, sqH, gptr,
                                            kDT, 1, outfield, bfA, sqA);
        cur = bufA;
    }
}
